// PointNet2_DA_28862180229719
// MI455X (gfx1250) — matrix operand order repaired and checked by reading, not compiled, not run on hardware
//
#include <hip/hip_runtime.h>
#include <hip/hip_bf16.h>
#include <stdint.h>
#include <math.h>

typedef __attribute__((ext_vector_type(16))) _Float16 v16h;
typedef __attribute__((ext_vector_type(8)))  _Float16 v8h;
typedef __attribute__((ext_vector_type(8)))  float    v8f;

namespace pn2 {

constexpr int   Bb   = 16;
constexpr int   Nn   = 4096;
constexpr int   NCls = 40;
constexpr float EPSf = 1e-5f;

constexpr int S1 = 512, NS1 = 32;   // sa1: 512 centroids, 32 neighbors
constexpr int S2 = 128, NS2 = 64;   // sa2: 128 centroids, 64 neighbors

constexpr int M1 = Bb * S1 * NS1;   // 262144 rows for sa1 MLP
constexpr int M2 = Bb * S2 * NS2;   // 131072 rows for sa2 MLP
constexpr int M3 = Bb * S2;         // 2048 rows for sa3 MLP

constexpr size_t alignup(size_t x) { return (x + 255) & ~size_t(255); }

// ---- workspace layout (bytes) ----
constexpr size_t OFF_PTS   = 0;                                         // B*N*3 f32
constexpr size_t OFF_FPS1  = alignup(OFF_PTS   + size_t(Bb)*Nn*3*4);    // B*S1 i32
constexpr size_t OFF_NX1   = alignup(OFF_FPS1  + size_t(Bb)*S1*4);      // B*S1*3 f32
constexpr size_t OFF_GI1   = alignup(OFF_NX1   + size_t(Bb)*S1*3*4);    // B*S1*NS1 i32
constexpr size_t OFF_X1    = alignup(OFF_GI1   + size_t(Bb)*S1*NS1*4);  // M1*32 f16
constexpr size_t OFF_W11   = alignup(OFF_X1    + size_t(M1)*32*2);      // 64*32 f16
constexpr size_t OFF_W12   = alignup(OFF_W11   + size_t(64)*32*2);      // 64*64
constexpr size_t OFF_W13   = alignup(OFF_W12   + size_t(64)*64*2);      // 128*64
constexpr size_t OFF_W21   = alignup(OFF_W13   + size_t(128)*64*2);     // 128*160
constexpr size_t OFF_W22   = alignup(OFF_W21   + size_t(128)*160*2);    // 128*128
constexpr size_t OFF_W23   = alignup(OFF_W22   + size_t(128)*128*2);    // 256*128
constexpr size_t OFF_W31   = alignup(OFF_W23   + size_t(256)*128*2);    // 256*288
constexpr size_t OFF_W32   = alignup(OFF_W31   + size_t(256)*288*2);    // 512*256
constexpr size_t OFF_W33   = alignup(OFF_W32   + size_t(512)*256*2);    // 1024*512
constexpr size_t OFF_BUFA  = alignup(OFF_W33   + size_t(1024)*512*2);   // M1*128 f16 max
constexpr size_t OFF_BUFB  = alignup(OFF_BUFA  + size_t(M1)*128*2);
constexpr size_t OFF_STAT  = alignup(OFF_BUFB  + size_t(M1)*128*2);     // 2*1024 f32
constexpr size_t OFF_SCL   = alignup(OFF_STAT  + size_t(2)*1024*4);     // 1024 f32
constexpr size_t OFF_SHF   = alignup(OFF_SCL   + size_t(1024)*4);       // 1024 f32
constexpr size_t OFF_L1P   = alignup(OFF_SHF   + size_t(1024)*4);       // B*S1*128 f16
constexpr size_t OFF_FPS2  = alignup(OFF_L1P   + size_t(Bb)*S1*128*2);  // B*S2 i32
constexpr size_t OFF_NX2   = alignup(OFF_FPS2  + size_t(Bb)*S2*4);      // B*S2*3 f32
constexpr size_t OFF_GI2   = alignup(OFF_NX2   + size_t(Bb)*S2*3*4);    // B*S2*NS2 i32
constexpr size_t OFF_X2    = alignup(OFF_GI2   + size_t(Bb)*S2*NS2*4);  // M2*160 f16
constexpr size_t OFF_L2P   = alignup(OFF_X2    + size_t(M2)*160*2);     // M3*256 f16
constexpr size_t OFF_X3    = alignup(OFF_L2P   + size_t(M3)*256*2);     // M3*288 f16
constexpr size_t OFF_XH    = alignup(OFF_X3    + size_t(M3)*288*2);     // 16*1024 f32
constexpr size_t OFF_Z1    = alignup(OFF_XH    + size_t(16)*1024*4);    // 16*512 f32
constexpr size_t OFF_A1    = alignup(OFF_Z1    + size_t(16)*512*4);
constexpr size_t OFF_Z2    = alignup(OFF_A1    + size_t(16)*512*4);     // 16*256
constexpr size_t OFF_A2    = alignup(OFF_Z2    + size_t(16)*256*4);
constexpr size_t OFF_Z3    = alignup(OFF_A2    + size_t(16)*256*4);     // 16*40

} // namespace pn2

using namespace pn2;

// ---------------------------------------------------------------------------
// (B,3,N) -> (B,N,3)
__global__ void k_transpose(const float* __restrict__ xyz, float* __restrict__ pts) {
  int i = blockIdx.x * blockDim.x + threadIdx.x;
  if (i >= Bb * Nn) return;
  int b = i / Nn, n = i - b * Nn;
  const float* src = xyz + (size_t)b * 3 * Nn;
  float* dst = pts + (size_t)i * 3;
  dst[0] = src[0 * Nn + n];
  dst[1] = src[1 * Nn + n];
  dst[2] = src[2 * Nn + n];
}

// ---------------------------------------------------------------------------
// Farthest point sampling: one block per batch, matches jax.lax.scan semantics
// (selected[0]=0, argmax with first-index tie-break).
__global__ __launch_bounds__(1024)
void k_fps(const float* __restrict__ pts, int n, int npoint, int* __restrict__ outIdx) {
  __shared__ float dist[4096];
  __shared__ float rv[1024];
  __shared__ int   ri[1024];
  const int b = blockIdx.x, t = threadIdx.x, T = blockDim.x;
  const float* p = pts + (size_t)b * n * 3;
  for (int i = t; i < n; i += T) dist[i] = 1e10f;
  __syncthreads();
  int farthest = 0;
  for (int it = 0; it < npoint; ++it) {
    if (t == 0) outIdx[b * npoint + it] = farthest;
    const float cx = p[farthest * 3 + 0];
    const float cy = p[farthest * 3 + 1];
    const float cz = p[farthest * 3 + 2];
    float bestv = -1.0f; int besti = n;
    for (int i = t; i < n; i += T) {
      float dx = p[i * 3 + 0] - cx;
      float dy = p[i * 3 + 1] - cy;
      float dz = p[i * 3 + 2] - cz;
      float d = dx * dx + dy * dy + dz * dz;
      float dm = fminf(dist[i], d);
      dist[i] = dm;
      if (dm > bestv) { bestv = dm; besti = i; }
    }
    rv[t] = bestv; ri[t] = besti;
    __syncthreads();
    for (int s = T >> 1; s > 0; s >>= 1) {
      if (t < s) {
        if (rv[t + s] > rv[t] || (rv[t + s] == rv[t] && ri[t + s] < ri[t])) {
          rv[t] = rv[t + s]; ri[t] = ri[t + s];
        }
      }
      __syncthreads();
    }
    farthest = ri[0];
    __syncthreads();
  }
}

// ---------------------------------------------------------------------------
__global__ void k_gather_xyz(const float* __restrict__ pts, const int* __restrict__ idx,
                             float* __restrict__ out, int n, int npoint) {
  int i = blockIdx.x * blockDim.x + threadIdx.x;
  if (i >= Bb * npoint) return;
  int b = i / npoint;
  int id = idx[i];
  const float* src = pts + ((size_t)b * n + id) * 3;
  float* dst = out + (size_t)i * 3;
  dst[0] = src[0]; dst[1] = src[1]; dst[2] = src[2];
}

// ---------------------------------------------------------------------------
// Ball query: one wave32 per centroid; first `nsample` in-radius points in
// ascending index order, padded with the first hit (matches sort+slice ref).
__global__ void k_ballquery(const float* __restrict__ pts, const float* __restrict__ centers,
                            int n, int S, int nsample, float r2, int* __restrict__ outIdx) {
  int wid = (int)((blockIdx.x * (size_t)blockDim.x + threadIdx.x) >> 5);
  int lane = threadIdx.x & 31;
  if (wid >= Bb * S) return;
  int b = wid / S;
  const float* p = pts + (size_t)b * n * 3;
  const float cx = centers[(size_t)wid * 3 + 0];
  const float cy = centers[(size_t)wid * 3 + 1];
  const float cz = centers[(size_t)wid * 3 + 2];
  int* out = outIdx + (size_t)wid * nsample;
  int count = 0;
  int firstIdx = 0;
  bool found = false;
  for (int j0 = 0; j0 < n && count < nsample; j0 += 32) {
    int j = j0 + lane;
    bool pred = false;
    if (j < n) {
      float dx = p[j * 3 + 0] - cx;
      float dy = p[j * 3 + 1] - cy;
      float dz = p[j * 3 + 2] - cz;
      pred = (dx * dx + dy * dy + dz * dz) <= r2;
    }
    unsigned long long mask = __ballot(pred);
    if (!found && mask) { firstIdx = j0 + ((int)__ffsll(mask) - 1); found = true; }
    if (pred) {
      int pos = count + (int)__popcll(mask & ((1ull << lane) - 1ull));
      if (pos < nsample) out[pos] = j;
    }
    count += (int)__popcll(mask);
  }
  if (count < nsample) {
    for (int pos = count + lane; pos < nsample; pos += 32) out[pos] = firstIdx;
  }
}

// ---------------------------------------------------------------------------
// Build sa1 input matrix: (M1, 32) f16, cols 0..2 = relative xyz, rest zero.
__global__ void k_buildX1(const float* __restrict__ pts, const float* __restrict__ nx,
                          const int* __restrict__ gidx, _Float16* __restrict__ X) {
  int i = blockIdx.x * blockDim.x + threadIdx.x;
  if (i >= M1) return;
  int b = i / (S1 * NS1);
  int s = (i / NS1) % S1;
  int id = gidx[i];
  const float* pp = pts + ((size_t)b * Nn + id) * 3;
  const float* c  = nx  + ((size_t)b * S1 + s) * 3;
  _Float16* row = X + (size_t)i * 32;
  row[0] = (_Float16)(pp[0] - c[0]);
  row[1] = (_Float16)(pp[1] - c[1]);
  row[2] = (_Float16)(pp[2] - c[2]);
  #pragma unroll
  for (int k = 3; k < 32; ++k) row[k] = (_Float16)0.0f;
}

// Build sa2 input: (M2, 160) f16, cols 0..2 rel xyz, 3..130 = l1 features, rest 0.
__global__ void k_buildX2(const float* __restrict__ l1xyz, const _Float16* __restrict__ l1p,
                          const float* __restrict__ nx2, const int* __restrict__ gidx,
                          _Float16* __restrict__ X) {
  int i = blockIdx.x * blockDim.x + threadIdx.x;
  if (i >= M2) return;
  int b = i / (S2 * NS2);
  int s = (i / NS2) % S2;
  int id = gidx[i]; // in [0, S1)
  const float* pp = l1xyz + ((size_t)b * S1 + id) * 3;
  const float* c  = nx2   + ((size_t)b * S2 + s)  * 3;
  _Float16* row = X + (size_t)i * 160;
  row[0] = (_Float16)(pp[0] - c[0]);
  row[1] = (_Float16)(pp[1] - c[1]);
  row[2] = (_Float16)(pp[2] - c[2]);
  const _Float16* f = l1p + ((size_t)b * S1 + id) * 128;
  for (int k = 0; k < 128; ++k) row[3 + k] = f[k];
  for (int k = 131; k < 160; ++k) row[k] = (_Float16)0.0f;
}

// Build sa3 input: (M3, 288) f16, cols 0..2 = l2 xyz (absolute), 3..258 = l2 feats.
__global__ void k_buildX3(const float* __restrict__ l2xyz, const _Float16* __restrict__ l2p,
                          _Float16* __restrict__ X) {
  int i = blockIdx.x * blockDim.x + threadIdx.x;
  if (i >= M3) return;
  _Float16* row = X + (size_t)i * 288;
  row[0] = (_Float16)l2xyz[(size_t)i * 3 + 0];
  row[1] = (_Float16)l2xyz[(size_t)i * 3 + 1];
  row[2] = (_Float16)l2xyz[(size_t)i * 3 + 2];
  const _Float16* f = l2p + (size_t)i * 256;
  for (int k = 0; k < 256; ++k) row[3 + k] = f[k];
  for (int k = 259; k < 288; ++k) row[k] = (_Float16)0.0f;
}

// ---------------------------------------------------------------------------
// Convert + zero-pad weights: (Cout, Cin) f32 -> (Cout, Kpad) f16.
__global__ void k_prepW(const float* __restrict__ w, _Float16* __restrict__ wh,
                        int Cout, int Cin, int Kpad) {
  int i = blockIdx.x * blockDim.x + threadIdx.x;
  if (i >= Cout * Kpad) return;
  int o = i / Kpad, k = i - o * Kpad;
  wh[i] = (_Float16)(k < Cin ? w[(size_t)o * Cin + k] : 0.0f);
}

// ---------------------------------------------------------------------------
// WMMA GEMM: Y(M,Cout) = act(X)(M,K) * Wh(Cout,K)^T + bias, f16 in / f32 acc /
// f16 out. Block layout: all 8 waves of a block share one 64-column chunk,
// whose 64xK f16 weight slab is staged in (dynamic) LDS once per block.
// Each wave computes a 16x64 output strip: A fragment loaded once per K-step
// (global), NT=4 B fragments read from LDS into DISTINCT registers, then 4
// back-to-back v_wmma ops. sched_barrier(0) pins the load/compute split so the
// scheduler cannot sink ds_loads between WMMAs (which would force per-WMMA
// s_wait_dscnt + WAR-hazard v_nops on shared B registers).
template <bool ACT, int NT>
__global__ __launch_bounds__(256)
void k_gemm_t(const _Float16* __restrict__ X, const _Float16* __restrict__ Wh,
              const float* __restrict__ bias,
              const float* __restrict__ actScale, const float* __restrict__ actShift,
              _Float16* __restrict__ Y, int M, int K, int Cout) {
  extern __shared__ _Float16 ldsW[];         // 64 * K halves
  const int chunksN = Cout / (16 * NT);
  const int ch = blockIdx.x % chunksN;       // column chunk for the whole block
  const int tm = (blockIdx.x / chunksN) * 8 + (threadIdx.x >> 5);
  const int lane = threadIdx.x & 31;

  // --- stage the 64xK weight slab into LDS (cooperative, 16B vectors) ---
  {
    const v8h* wsrc = (const v8h*)(Wh + (size_t)(ch * 16 * NT) * K);
    v8h* wdst = (v8h*)ldsW;
    const int nvec = (16 * NT * K) / 8;
    for (int idx = threadIdx.x; idx < nvec; idx += 256) wdst[idx] = wsrc[idx];
  }
  __syncthreads();

  const int tilesM = M >> 4;
  if (tm >= tilesM) return;                  // wave-uniform exit (EXEC stays full)

  const int mrow = tm * 16 + (lane & 15);
  const int ksel = lane >> 4;                // 0/1: which K-subchunk this lane holds
  const _Float16* xr = X + (size_t)mrow * K;
  const _Float16* bl = ldsW + (size_t)(lane & 15) * K;  // column (lane&15) base
  v8f acc[NT];
  v8f zero = {};
  #pragma unroll
  for (int t = 0; t < NT; ++t) acc[t] = zero;

  for (int k0 = 0; k0 < K; k0 += 32) {
    __builtin_prefetch(xr + k0 + 32, 0, 3);  // next A chunk, near-temporal
    // A fragment (16x32 f16): halves 0..15 -> K = k0+ksel*16+j (contiguous)
    v8h a0 = *(const v8h*)(xr + k0 + ksel * 16);
    v8h a1 = *(const v8h*)(xr + k0 + ksel * 16 + 8);
    if constexpr (ACT) {
      v8f s0 = *(const v8f*)(actScale + k0 + ksel * 16);
      v8f t0 = *(const v8f*)(actShift + k0 + ksel * 16);
      v8f s1 = *(const v8f*)(actScale + k0 + ksel * 16 + 8);
      v8f t1 = *(const v8f*)(actShift + k0 + ksel * 16 + 8);
      #pragma unroll
      for (int j = 0; j < 8; ++j) {
        a0[j] = (_Float16)fmaxf((float)a0[j] * s0[j] + t0[j], 0.0f);
        a1[j] = (_Float16)fmaxf((float)a1[j] * s1[j] + t1[j], 0.0f);
      }
    }
    v16h a;
    #pragma unroll
    for (int j = 0; j < 8; ++j) { a[j] = a0[j]; a[8 + j] = a1[j]; }
    // load ALL NT B fragments from LDS first (distinct registers) ...
    v16h bf[NT];
    #pragma unroll
    for (int t = 0; t < NT; ++t) {
      bf[t] = *(const v16h*)(bl + (size_t)t * 16 * K + k0 + ksel * 16);
    }
    __builtin_amdgcn_sched_barrier(0);  // keep loads above, WMMAs below
    #pragma unroll
    for (int t = 0; t < NT; ++t) {
      acc[t] = __builtin_amdgcn_wmma_f32_16x16x32_f16(false, a, false, bf[t],
                                                      (short)0, acc[t], false, false);
    }
    __builtin_amdgcn_sched_barrier(0);  // keep next iteration's work below
  }
  const int rbase = tm * 16 + ksel * 8;
  #pragma unroll
  for (int t = 0; t < NT; ++t) {
    const int nc = ch * (16 * NT) + t * 16 + (lane & 15);
    const float bc = bias[nc];
    #pragma unroll
    for (int j = 0; j < 8; ++j) {
      Y[(size_t)(rbase + j) * Cout + nc] = (_Float16)(acc[t][j] + bc);
    }
  }
}

// ---------------------------------------------------------------------------
// Per-channel sum / sum-of-squares over M rows (bn statistics).
// C is a power of two (lgC = log2(C)); each thread owns one channel column
// (coalesced across lanes), accumulates in registers, one LDS atomic at end.
// Requires gridDim.x*blockDim.x to be a multiple of C.
__global__ void k_stats(const _Float16* __restrict__ Y, int Mrows, int C, int lgC,
                        float* __restrict__ acc) {
  __shared__ float s1[1024];
  __shared__ float s2[1024];
  int t = threadIdx.x;
  for (int c = t; c < C; c += blockDim.x) { s1[c] = 0.0f; s2[c] = 0.0f; }
  __syncthreads();
  int gid = blockIdx.x * blockDim.x + t;
  int gsize = gridDim.x * blockDim.x;
  int c = gid & (C - 1);
  int r0 = gid >> lgC;
  int rstep = gsize >> lgC;
  float sl = 0.0f, sq = 0.0f;
  for (int r = r0; r < Mrows; r += rstep) {
    float v = (float)Y[(size_t)r * C + c];
    sl += v; sq += v * v;
  }
  atomicAdd(&s1[c], sl);
  atomicAdd(&s2[c], sq);
  __syncthreads();
  for (int cc = t; cc < C; cc += blockDim.x) {
    atomicAdd(&acc[cc], s1[cc]);
    atomicAdd(&acc[C + cc], s2[cc]);
  }
}

__global__ void k_finalize(const float* __restrict__ acc, const float* __restrict__ g,
                           const float* __restrict__ be, float invM, int C,
                           float* __restrict__ scale, float* __restrict__ shift) {
  int c = blockIdx.x * blockDim.x + threadIdx.x;
  if (c >= C) return;
  float mean = acc[c] * invM;
  float var  = acc[C + c] * invM - mean * mean;
  float a = g[c] * rsqrtf(var + EPSf);
  scale[c] = a;
  shift[c] = be[c] - mean * a;
}

// ---------------------------------------------------------------------------
// Apply bn+relu then max over KS contiguous rows per pooled row.
__global__ void k_maxpool(const _Float16* __restrict__ Y, const float* __restrict__ scale,
                          const float* __restrict__ shift, int rows, int C, int KS,
                          _Float16* __restrict__ outH, float* __restrict__ outF,
                          int writeF32) {
  int i = blockIdx.x * blockDim.x + threadIdx.x;
  if (i >= rows * C) return;
  int r = i / C, c = i - r * C;
  float a = scale[c], sh = shift[c];
  const _Float16* base = Y + (size_t)r * KS * C + c;
  float m = -3.0e38f;
  for (int k = 0; k < KS; ++k) {
    float v = fmaxf((float)base[(size_t)k * C] * a + sh, 0.0f);
    m = fmaxf(m, v);
  }
  if (writeF32) outF[i] = m;
  else          outH[i] = (_Float16)m;
}

// ---------------------------------------------------------------------------
// Head: plain f32 FC (tiny), batch-bn+relu, log-softmax.
__global__ void k_fc(const float* __restrict__ X, const float* __restrict__ W,
                     const float* __restrict__ bias, float* __restrict__ Z,
                     int M, int Kd, int O) {
  int i = blockIdx.x * blockDim.x + threadIdx.x;
  if (i >= M * O) return;
  int m = i / O, o = i - m * O;
  const float* x = X + (size_t)m * Kd;
  const float* w = W + (size_t)o * Kd;
  float s = bias[o];
  for (int k = 0; k < Kd; ++k) s = fmaf(x[k], w[k], s);
  Z[i] = s;
}

__global__ void k_bnrelu_batch(const float* __restrict__ Z, const float* __restrict__ g,
                               const float* __restrict__ be, float* __restrict__ Xo,
                               int M, int C) {
  int c = blockIdx.x * blockDim.x + threadIdx.x;
  if (c >= C) return;
  float mean = 0.0f;
  for (int m = 0; m < M; ++m) mean += Z[(size_t)m * C + c];
  mean /= (float)M;
  float var = 0.0f;
  for (int m = 0; m < M; ++m) { float d = Z[(size_t)m * C + c] - mean; var += d * d; }
  var /= (float)M;
  float a = g[c] * rsqrtf(var + EPSf);
  float sh = be[c] - mean * a;
  for (int m = 0; m < M; ++m)
    Xo[(size_t)m * C + c] = fmaxf(a * Z[(size_t)m * C + c] + sh, 0.0f);
}

__global__ void k_logsoftmax(const float* __restrict__ Z, float* __restrict__ out,
                             int M, int C) {
  int m = blockIdx.x * blockDim.x + threadIdx.x;
  if (m >= M) return;
  const float* z = Z + (size_t)m * C;
  float mx = -3.0e38f;
  for (int c = 0; c < C; ++c) mx = fmaxf(mx, z[c]);
  float s = 0.0f;
  for (int c = 0; c < C; ++c) s += expf(z[c] - mx);
  float lse = mx + logf(s);
  for (int c = 0; c < C; ++c) out[(size_t)m * C + c] = z[c] - lse;
}

// ---------------------------------------------------------------------------
static inline void launch_gemm(const _Float16* X, const _Float16* Wh, const float* bias,
                               const float* scl, const float* shf, _Float16* Y,
                               int M, int K, int Cout, hipStream_t st) {
  constexpr int NT = 4;                      // all Cout here are multiples of 64
  int chunksN = Cout / (16 * NT);
  int mb = (M / 16 + 7) / 8;
  int blocks = chunksN * mb;
  size_t ldsBytes = (size_t)(16 * NT) * K * sizeof(_Float16);  // <= 64 KB (K<=512)
  if (scl) {
    k_gemm_t<true, NT><<<blocks, 256, ldsBytes, st>>>(X, Wh, bias, scl, shf, Y, M, K, Cout);
  } else {
    k_gemm_t<false, NT><<<blocks, 256, ldsBytes, st>>>(X, Wh, bias, nullptr, nullptr, Y, M, K, Cout);
  }
}

static inline int cdiv(long long a, int b) { return (int)((a + b - 1) / b); }

extern "C" void kernel_launch(void* const* d_in, const int* in_sizes, int n_in,
                              void* d_out, int out_size, void* d_ws, size_t ws_size,
                              hipStream_t stream) {
  (void)in_sizes; (void)n_in; (void)out_size; (void)ws_size;
  char* ws = (char*)d_ws;

  // --- inputs (setup_inputs dict order, params in insertion order) ---
  const float* xyz = (const float*)d_in[0];
  // sa layers: (w, bias, g, be) x3 per stage
  const float* w11 = (const float*)d_in[1];  const float* b11 = (const float*)d_in[2];
  const float* g11 = (const float*)d_in[3];  const float* e11 = (const float*)d_in[4];
  const float* w12 = (const float*)d_in[5];  const float* b12 = (const float*)d_in[6];
  const float* g12 = (const float*)d_in[7];  const float* e12 = (const float*)d_in[8];
  const float* w13 = (const float*)d_in[9];  const float* b13 = (const float*)d_in[10];
  const float* g13 = (const float*)d_in[11]; const float* e13 = (const float*)d_in[12];
  const float* w21 = (const float*)d_in[13]; const float* b21 = (const float*)d_in[14];
  const float* g21 = (const float*)d_in[15]; const float* e21 = (const float*)d_in[16];
  const float* w22 = (const float*)d_in[17]; const float* b22 = (const float*)d_in[18];
  const float* g22 = (const float*)d_in[19]; const float* e22 = (const float*)d_in[20];
  const float* w23 = (const float*)d_in[21]; const float* b23 = (const float*)d_in[22];
  const float* g23 = (const float*)d_in[23]; const float* e23 = (const float*)d_in[24];
  const float* w31 = (const float*)d_in[25]; const float* b31 = (const float*)d_in[26];
  const float* g31 = (const float*)d_in[27]; const float* e31 = (const float*)d_in[28];
  const float* w32 = (const float*)d_in[29]; const float* b32 = (const float*)d_in[30];
  const float* g32 = (const float*)d_in[31]; const float* e32 = (const float*)d_in[32];
  const float* w33 = (const float*)d_in[33]; const float* b33 = (const float*)d_in[34];
  const float* g33 = (const float*)d_in[35]; const float* e33 = (const float*)d_in[36];
  const float* wf1 = (const float*)d_in[37]; const float* bf1 = (const float*)d_in[38];
  const float* wf2 = (const float*)d_in[39]; const float* bf2 = (const float*)d_in[40];
  const float* wf3 = (const float*)d_in[41]; const float* bf3 = (const float*)d_in[42];
  const float* gn1 = (const float*)d_in[43]; const float* en1 = (const float*)d_in[44];
  const float* gn2 = (const float*)d_in[45]; const float* en2 = (const float*)d_in[46];

  // --- workspace views ---
  float*    pts  = (float*)(ws + OFF_PTS);
  int*      fps1 = (int*)  (ws + OFF_FPS1);
  float*    nx1  = (float*)(ws + OFF_NX1);
  int*      gi1  = (int*)  (ws + OFF_GI1);
  _Float16* X1   = (_Float16*)(ws + OFF_X1);
  _Float16* W11h = (_Float16*)(ws + OFF_W11);
  _Float16* W12h = (_Float16*)(ws + OFF_W12);
  _Float16* W13h = (_Float16*)(ws + OFF_W13);
  _Float16* W21h = (_Float16*)(ws + OFF_W21);
  _Float16* W22h = (_Float16*)(ws + OFF_W22);
  _Float16* W23h = (_Float16*)(ws + OFF_W23);
  _Float16* W31h = (_Float16*)(ws + OFF_W31);
  _Float16* W32h = (_Float16*)(ws + OFF_W32);
  _Float16* W33h = (_Float16*)(ws + OFF_W33);
  _Float16* bufA = (_Float16*)(ws + OFF_BUFA);
  _Float16* bufB = (_Float16*)(ws + OFF_BUFB);
  float*    stat = (float*)(ws + OFF_STAT);
  float*    scl  = (float*)(ws + OFF_SCL);
  float*    shf  = (float*)(ws + OFF_SHF);
  _Float16* l1p  = (_Float16*)(ws + OFF_L1P);
  int*      fps2 = (int*)  (ws + OFF_FPS2);
  float*    nx2  = (float*)(ws + OFF_NX2);
  int*      gi2  = (int*)  (ws + OFF_GI2);
  _Float16* X2   = (_Float16*)(ws + OFF_X2);
  _Float16* l2p  = (_Float16*)(ws + OFF_L2P);
  _Float16* X3   = (_Float16*)(ws + OFF_X3);
  float*    xh   = (float*)(ws + OFF_XH);
  float*    z1   = (float*)(ws + OFF_Z1);
  float*    a1   = (float*)(ws + OFF_A1);
  float*    z2   = (float*)(ws + OFF_Z2);
  float*    a2   = (float*)(ws + OFF_A2);
  float*    z3   = (float*)(ws + OFF_Z3);
  float*    out  = (float*)d_out;

  const size_t statBytes = 2 * 1024 * 4;

  // --- weight conversion (f32 -> padded f16) ---
  k_prepW<<<cdiv(64 * 32, 256), 256, 0, stream>>>(w11, W11h, 64, 3, 32);
  k_prepW<<<cdiv(64 * 64, 256), 256, 0, stream>>>(w12, W12h, 64, 64, 64);
  k_prepW<<<cdiv(128 * 64, 256), 256, 0, stream>>>(w13, W13h, 128, 64, 64);
  k_prepW<<<cdiv(128 * 160, 256), 256, 0, stream>>>(w21, W21h, 128, 131, 160);
  k_prepW<<<cdiv(128 * 128, 256), 256, 0, stream>>>(w22, W22h, 128, 128, 128);
  k_prepW<<<cdiv(256 * 128, 256), 256, 0, stream>>>(w23, W23h, 256, 128, 128);
  k_prepW<<<cdiv(256 * 288, 256), 256, 0, stream>>>(w31, W31h, 256, 259, 288);
  k_prepW<<<cdiv(512 * 256, 256), 256, 0, stream>>>(w32, W32h, 512, 256, 256);
  k_prepW<<<cdiv(1024 * 512, 256), 256, 0, stream>>>(w33, W33h, 1024, 512, 512);

  // --- stage 0: transpose ---
  k_transpose<<<cdiv((long long)Bb * Nn, 256), 256, 0, stream>>>(xyz, pts);

  // --- sa1: FPS + ball query + group ---
  k_fps<<<Bb, 1024, 0, stream>>>(pts, Nn, S1, fps1);
  k_gather_xyz<<<cdiv(Bb * S1, 256), 256, 0, stream>>>(pts, fps1, nx1, Nn, S1);
  k_ballquery<<<cdiv((long long)Bb * S1 * 32, 256), 256, 0, stream>>>(
      pts, nx1, Nn, S1, NS1, 0.2f * 0.2f, gi1);
  k_buildX1<<<cdiv(M1, 256), 256, 0, stream>>>(pts, nx1, gi1, X1);

  // --- sa1 MLP [64,64,128] ---
  launch_gemm(X1, W11h, b11, nullptr, nullptr, bufA, M1, 32, 64, stream);
  hipMemsetAsync(stat, 0, statBytes, stream);
  k_stats<<<512, 256, 0, stream>>>(bufA, M1, 64, 6, stat);
  k_finalize<<<1, 64, 0, stream>>>(stat, g11, e11, 1.0f / M1, 64, scl, shf);
  launch_gemm(bufA, W12h, b12, scl, shf, bufB, M1, 64, 64, stream);
  hipMemsetAsync(stat, 0, statBytes, stream);
  k_stats<<<512, 256, 0, stream>>>(bufB, M1, 64, 6, stat);
  k_finalize<<<1, 64, 0, stream>>>(stat, g12, e12, 1.0f / M1, 64, scl, shf);
  launch_gemm(bufB, W13h, b13, scl, shf, bufA, M1, 64, 128, stream);
  hipMemsetAsync(stat, 0, statBytes, stream);
  k_stats<<<512, 256, 0, stream>>>(bufA, M1, 128, 7, stat);
  k_finalize<<<1, 128, 0, stream>>>(stat, g13, e13, 1.0f / M1, 128, scl, shf);
  k_maxpool<<<cdiv((long long)Bb * S1 * 128, 256), 256, 0, stream>>>(
      bufA, scl, shf, Bb * S1, 128, NS1, l1p, nullptr, 0);

  // --- sa2: FPS + ball query + group ---
  k_fps<<<Bb, 1024, 0, stream>>>(nx1, S1, S2, fps2);
  k_gather_xyz<<<cdiv(Bb * S2, 256), 256, 0, stream>>>(nx1, fps2, nx2, S1, S2);
  k_ballquery<<<cdiv((long long)Bb * S2 * 32, 256), 256, 0, stream>>>(
      nx1, nx2, S1, S2, NS2, 0.4f * 0.4f, gi2);
  k_buildX2<<<cdiv(M2, 256), 256, 0, stream>>>(nx1, l1p, nx2, gi2, X2);

  // --- sa2 MLP [128,128,256] ---
  launch_gemm(X2, W21h, b21, nullptr, nullptr, bufA, M2, 160, 128, stream);
  hipMemsetAsync(stat, 0, statBytes, stream);
  k_stats<<<512, 256, 0, stream>>>(bufA, M2, 128, 7, stat);
  k_finalize<<<1, 128, 0, stream>>>(stat, g21, e21, 1.0f / M2, 128, scl, shf);
  launch_gemm(bufA, W22h, b22, scl, shf, bufB, M2, 128, 128, stream);
  hipMemsetAsync(stat, 0, statBytes, stream);
  k_stats<<<512, 256, 0, stream>>>(bufB, M2, 128, 7, stat);
  k_finalize<<<1, 128, 0, stream>>>(stat, g22, e22, 1.0f / M2, 128, scl, shf);
  launch_gemm(bufB, W23h, b23, scl, shf, bufA, M2, 128, 256, stream);
  hipMemsetAsync(stat, 0, statBytes, stream);
  k_stats<<<512, 256, 0, stream>>>(bufA, M2, 256, 8, stat);
  k_finalize<<<1, 256, 0, stream>>>(stat, g23, e23, 1.0f / M2, 256, scl, shf);
  k_maxpool<<<cdiv((long long)M3 * 256, 256), 256, 0, stream>>>(
      bufA, scl, shf, M3, 256, NS2, l2p, nullptr, 0);

  // --- sa3: group-all ---
  k_buildX3<<<cdiv(M3, 256), 256, 0, stream>>>(nx2, l2p, X3);

  // --- sa3 MLP [256,512,1024] ---
  launch_gemm(X3, W31h, b31, nullptr, nullptr, bufA, M3, 288, 256, stream);
  hipMemsetAsync(stat, 0, statBytes, stream);
  k_stats<<<256, 256, 0, stream>>>(bufA, M3, 256, 8, stat);
  k_finalize<<<1, 256, 0, stream>>>(stat, g31, e31, 1.0f / M3, 256, scl, shf);
  launch_gemm(bufA, W32h, b32, scl, shf, bufB, M3, 256, 512, stream);
  hipMemsetAsync(stat, 0, statBytes, stream);
  k_stats<<<256, 256, 0, stream>>>(bufB, M3, 512, 9, stat);
  k_finalize<<<2, 256, 0, stream>>>(stat, g32, e32, 1.0f / M3, 512, scl, shf);
  launch_gemm(bufB, W33h, b33, scl, shf, bufA, M3, 512, 1024, stream);
  hipMemsetAsync(stat, 0, statBytes, stream);
  k_stats<<<256, 256, 0, stream>>>(bufA, M3, 1024, 10, stat);
  k_finalize<<<4, 256, 0, stream>>>(stat, g33, e33, 1.0f / M3, 1024, scl, shf);
  k_maxpool<<<cdiv((long long)Bb * 1024, 256), 256, 0, stream>>>(
      bufA, scl, shf, Bb, 1024, S2, nullptr, xh, 1);

  // --- FC head ---
  k_fc<<<cdiv(16 * 512, 256), 256, 0, stream>>>(xh, wf1, bf1, z1, 16, 1024, 512);
  k_bnrelu_batch<<<cdiv(512, 256), 256, 0, stream>>>(z1, gn1, en1, a1, 16, 512);
  k_fc<<<cdiv(16 * 256, 256), 256, 0, stream>>>(a1, wf2, bf2, z2, 16, 512, 256);
  k_bnrelu_batch<<<1, 256, 0, stream>>>(z2, gn2, en2, a2, 16, 256);
  k_fc<<<cdiv(16 * NCls, 256), 256, 0, stream>>>(a2, wf3, bf3, z3, 16, 256, NCls);
  k_logsoftmax<<<1, 16, 0, stream>>>(z3, out, 16, NCls);
}
